// Attention_54614804136506
// MI455X (gfx1250) — matrix operand order repaired and checked by reading, not compiled, not run on hardware
//
#include <hip/hip_runtime.h>
#include <hip/hip_bf16.h>

typedef __attribute__((ext_vector_type(16))) __bf16 bf16x16;
typedef __attribute__((ext_vector_type(8)))  __bf16 bf16x8;
typedef __attribute__((ext_vector_type(8)))  float  f32x8;

#define NTOK   256
#define INDIM  768
#define DHEAD  64
#define NBATCH 256
#define SAWIN  32

// ---------------------------------------------------------------------------
// WMMA helper: D = A(16x32 bf16) * B(32x16 bf16) + C(16x16 f32)
// ---------------------------------------------------------------------------
__device__ __forceinline__ f32x8 wmma_bf16(bf16x16 a, bf16x16 b, f32x8 c) {
    return __builtin_amdgcn_wmma_f32_16x16x32_bf16(
        /*neg_a=*/false, a, /*neg_b=*/false, b,
        /*c_mod=*/(short)0, c, /*reuse_a=*/false, /*reuse_b=*/false);
}

// A fragment (16x32, 16-bit): lane L holds row M=L&15; K = (L>>4)*16 + 0..15
__device__ __forceinline__ bf16x16 load_a_bf16(const __bf16* A, int lda) {
    int lane = threadIdx.x & 31;
    int m    = lane & 15;
    int kb   = (lane >> 4) << 4;           // 0 or 16
    bf16x8 lo = *(const bf16x8*)(A + (size_t)m * lda + kb);
    bf16x8 hi = *(const bf16x8*)(A + (size_t)m * lda + kb + 8);
    bf16x16 f;
#pragma unroll
    for (int i = 0; i < 8; ++i) { f[i] = lo[i]; f[8 + i] = hi[i]; }
    return f;
}

// A fragment from fp32 source, converting to bf16 on the fly
__device__ __forceinline__ bf16x16 load_a_f32(const float* A, int lda) {
    int lane = threadIdx.x & 31;
    int m    = lane & 15;
    int kb   = (lane >> 4) << 4;
    const float* p = A + (size_t)m * lda + kb;
    float4 a0 = *(const float4*)(p);
    float4 a1 = *(const float4*)(p + 4);
    float4 a2 = *(const float4*)(p + 8);
    float4 a3 = *(const float4*)(p + 12);
    bf16x16 f;
    f[0]=(__bf16)a0.x; f[1]=(__bf16)a0.y; f[2]=(__bf16)a0.z; f[3]=(__bf16)a0.w;
    f[4]=(__bf16)a1.x; f[5]=(__bf16)a1.y; f[6]=(__bf16)a1.z; f[7]=(__bf16)a1.w;
    f[8]=(__bf16)a2.x; f[9]=(__bf16)a2.y; f[10]=(__bf16)a2.z; f[11]=(__bf16)a2.w;
    f[12]=(__bf16)a3.x; f[13]=(__bf16)a3.y; f[14]=(__bf16)a3.z; f[15]=(__bf16)a3.w;
    return f;
}

// B fragment (32x16, 16-bit): lane L holds col N=L&15; K = (L>>4)*16 + 0..15.
// Bt is stored transposed (N-major, contiguous along K); pointer at (n0,k0).
__device__ __forceinline__ bf16x16 load_b_bf16(const __bf16* Bt, int ldb) {
    int lane = threadIdx.x & 31;
    int n    = lane & 15;
    int kb   = (lane >> 4) << 4;           // 0 or 16
    return *(const bf16x16*)(Bt + (size_t)n * ldb + kb);
}

// ---------------------------------------------------------------------------
// prep: transpose + fp32->bf16 convert the two weight matrices
// ---------------------------------------------------------------------------
__global__ __launch_bounds__(256) void prep_kernel(const float* __restrict__ wqkv,
                                                   const float* __restrict__ wout,
                                                   __bf16* __restrict__ wqt,
                                                   __bf16* __restrict__ wot) {
    int i = blockIdx.x * blockDim.x + threadIdx.x;
    if (i < INDIM * 192) {                     // wqkv (768 x 192) -> wqt (192 x 768)
        int d = i / 192, e = i % 192;
        wqt[(size_t)e * INDIM + d] = (__bf16)wqkv[i];
    }
    if (i < DHEAD * INDIM) {                   // wout (64 x 768) -> wot (768 x 64)
        int h = i / INDIM, d = i % INDIM;
        wot[(size_t)d * DHEAD + h] = (__bf16)wout[i];
    }
}

// ---------------------------------------------------------------------------
// Kernel A: qkv = x @ w_qkv  (65536x768 @ 768x192)
// One wave per 32-row tile (2 M sub-tiles x 12 N tiles); each B fragment
// feeds two WMMAs, halving L2 weight traffic vs. M=16 tiling.
// Writes q,k row-major bf16 (65536x64) and v transposed per batch (64x256).
// ---------------------------------------------------------------------------
__global__ __launch_bounds__(256) void qkv_kernel(const float* __restrict__ x,
                                                  const __bf16* __restrict__ wqt,
                                                  __bf16* __restrict__ qws,
                                                  __bf16* __restrict__ kws,
                                                  __bf16* __restrict__ vtws) {
    int gw = (blockIdx.x * blockDim.x + threadIdx.x) >> 5;   // 0..2047
    size_t row0 = (size_t)gw * 32;

    f32x8 acc0[12] = {};
    f32x8 acc1[12] = {};
    for (int k0 = 0; k0 < INDIM; k0 += 32) {
        bf16x16 a0 = load_a_f32(x + row0 * INDIM + k0, INDIM);
        bf16x16 a1 = load_a_f32(x + (row0 + 16) * INDIM + k0, INDIM);
#pragma unroll
        for (int n = 0; n < 12; ++n) {
            bf16x16 bw = load_b_bf16(wqt + (size_t)(n * 16) * INDIM + k0, INDIM);
            acc0[n] = wmma_bf16(a0, bw, acc0[n]);
            acc1[n] = wmma_bf16(a1, bw, acc1[n]);
        }
    }

    int lane  = threadIdx.x & 31;
    int col   = lane & 15;
    int rbase = (lane >> 4) << 3;
    int b     = (int)(row0 >> 8);
    int tloc  = (int)(row0 & 255);

#pragma unroll
    for (int mi = 0; mi < 2; ++mi) {
        const f32x8* acc = mi ? acc1 : acc0;
        size_t rb = row0 + (size_t)mi * 16;
        // q: columns 0..63
#pragma unroll
        for (int n = 0; n < 4; ++n)
#pragma unroll
            for (int r = 0; r < 8; ++r)
                qws[(rb + rbase + r) * DHEAD + n * 16 + col] = (__bf16)acc[n][r];
        // k: columns 64..127
#pragma unroll
        for (int n = 4; n < 8; ++n)
#pragma unroll
            for (int r = 0; r < 8; ++r)
                kws[(rb + rbase + r) * DHEAD + (n - 4) * 16 + col] = (__bf16)acc[n][r];
        // v: columns 128..191, stored transposed per batch: vt[b][h][token]
#pragma unroll
        for (int n = 8; n < 12; ++n)
#pragma unroll
            for (int r = 0; r < 8; ++r)
                vtws[(size_t)b * DHEAD * NTOK + (size_t)((n - 8) * 16 + col) * NTOK
                     + tloc + mi * 16 + rbase + r] = (__bf16)acc[n][r];
    }
}

// ---------------------------------------------------------------------------
// Kernel B: banded attention, one wave per (batch, 16-query block)
// ---------------------------------------------------------------------------
__global__ __launch_bounds__(256) void attn_kernel(const __bf16* __restrict__ qws,
                                                   const __bf16* __restrict__ kws,
                                                   const __bf16* __restrict__ vtws,
                                                   __bf16* __restrict__ aout) {
    __shared__ __align__(16) __bf16 pbuf[8][16][96];   // per-wave P tile (16 x 96)

    int w    = threadIdx.x >> 5;
    int lane = threadIdx.x & 31;
    int gw   = blockIdx.x * 8 + w;
    int b    = gw >> 4;
    int qb   = gw & 15;
    int q0   = qb * 16;

    const __bf16* Q  = qws  + (size_t)b * NTOK * DHEAD;
    const __bf16* K  = kws  + (size_t)b * NTOK * DHEAD;
    const __bf16* VT = vtws + (size_t)b * DHEAD * NTOK;

    bf16x16 qa0 = load_a_bf16(Q + (size_t)q0 * DHEAD + 0,  DHEAD);
    bf16x16 qa1 = load_a_bf16(Q + (size_t)q0 * DHEAD + 32, DHEAD);

    const float scale = 0.125f;   // 64^-0.5
    int col   = lane & 15;
    int rbase = (lane >> 4) << 3;
    int t0    = qb - 2;

    // S = Q @ K^T over 5 key tiles, with band mask
    float s[5][8];
#pragma unroll
    for (int t = 0; t < 5; ++t) {
        int kt = t0 + t;
        if (kt < 0 || kt > 15) {
#pragma unroll
            for (int r = 0; r < 8; ++r) s[t][r] = -1e30f;
            continue;
        }
        f32x8 acc = {};
        bf16x16 kb0 = load_b_bf16(K + (size_t)(kt * 16) * DHEAD + 0,  DHEAD);
        acc = wmma_bf16(qa0, kb0, acc);
        bf16x16 kb1 = load_b_bf16(K + (size_t)(kt * 16) * DHEAD + 32, DHEAD);
        acc = wmma_bf16(qa1, kb1, acc);
        int key = kt * 16 + col;
#pragma unroll
        for (int r = 0; r < 8; ++r) {
            int qq = q0 + rbase + r;
            int d  = qq - key; d = d < 0 ? -d : d;
            s[t][r] = (d < SAWIN) ? acc[r] * scale : -1e30f;
        }
    }

    // row softmax: rows live in 16-lane halves; reduce with xor-shuffles 1,2,4,8
    float rinv[8];
#pragma unroll
    for (int r = 0; r < 8; ++r) {
        float m = s[0][r];
#pragma unroll
        for (int t = 1; t < 5; ++t) m = fmaxf(m, s[t][r]);
        m = fmaxf(m, __shfl_xor(m, 1, 32));
        m = fmaxf(m, __shfl_xor(m, 2, 32));
        m = fmaxf(m, __shfl_xor(m, 4, 32));
        m = fmaxf(m, __shfl_xor(m, 8, 32));
        float sum = 0.f;
#pragma unroll
        for (int t = 0; t < 5; ++t) { s[t][r] = __expf(s[t][r] - m); sum += s[t][r]; }
        sum += __shfl_xor(sum, 1, 32);
        sum += __shfl_xor(sum, 2, 32);
        sum += __shfl_xor(sum, 4, 32);
        sum += __shfl_xor(sum, 8, 32);
        rinv[r] = 1.0f / sum;
    }

    // write P (C-layout) into LDS as a row-major 16x96 bf16 tile, pad cols 80..95
#pragma unroll
    for (int t = 0; t < 5; ++t)
#pragma unroll
        for (int r = 0; r < 8; ++r)
            pbuf[w][rbase + r][t * 16 + col] = (__bf16)(s[t][r] * rinv[r]);
#pragma unroll
    for (int r = 0; r < 8; ++r)
        pbuf[w][rbase + r][80 + col] = (__bf16)0.0f;
    // same-wave LDS RAW: compiler inserts s_wait_dscnt; no barrier needed

    // O = P @ V   (16x96 @ 96x64), V read from transposed layout
    f32x8 o[4] = {};
    int kbase = q0 - 32;
#pragma unroll
    for (int kk = 0; kk < 3; ++kk) {
        bf16x16 pa = load_a_bf16(&pbuf[w][0][kk * 32], 96);
#pragma unroll
        for (int h = 0; h < 4; ++h) {
            int ks = kbase + kk * 32 + ((lane >> 4) << 4);
            ks = ks < 0 ? 0 : (ks > NTOK - 16 ? NTOK - 16 : ks);   // P==0 there
            bf16x16 vb = *(const bf16x16*)(VT + (size_t)(h * 16 + col) * NTOK + ks);
            o[h] = wmma_bf16(pa, vb, o[h]);
        }
    }

#pragma unroll
    for (int h = 0; h < 4; ++h)
#pragma unroll
        for (int r = 0; r < 8; ++r)
            aout[(size_t)(b * NTOK + q0 + rbase + r) * DHEAD + h * 16 + col]
                = (__bf16)o[h][r];
}

// ---------------------------------------------------------------------------
// Kernel C: out = attn_out @ w_out + b_out  (65536x64 @ 64x768)
// One wave per 64-row x 64-col tile: A frags held in registers, each B
// fragment feeds 4 WMMAs (4x less weight traffic than M=16 tiling).
// ---------------------------------------------------------------------------
__global__ __launch_bounds__(256) void out_kernel(const __bf16* __restrict__ aout,
                                                  const __bf16* __restrict__ wot,
                                                  const float* __restrict__ bout,
                                                  float* __restrict__ out) {
    int gw  = (blockIdx.x * blockDim.x + threadIdx.x) >> 5;  // 0..12287
    int mt  = gw / 12, cg = gw % 12;
    size_t row0 = (size_t)mt * 64;
    int col0 = cg * 64;

    bf16x16 a[4][2];
#pragma unroll
    for (int mi = 0; mi < 4; ++mi) {
        a[mi][0] = load_a_bf16(aout + (row0 + mi * 16) * DHEAD + 0,  DHEAD);
        a[mi][1] = load_a_bf16(aout + (row0 + mi * 16) * DHEAD + 32, DHEAD);
    }

    int lane  = threadIdx.x & 31;
    int col   = lane & 15;
    int rbase = (lane >> 4) << 3;

#pragma unroll
    for (int n = 0; n < 4; ++n) {
        bf16x16 b0 = load_b_bf16(wot + (size_t)(col0 + n * 16) * DHEAD + 0,  DHEAD);
        bf16x16 b1 = load_b_bf16(wot + (size_t)(col0 + n * 16) * DHEAD + 32, DHEAD);
        float bias = bout[col0 + n * 16 + col];
        f32x8 acc[4] = {};
#pragma unroll
        for (int mi = 0; mi < 4; ++mi) {
            acc[mi] = wmma_bf16(a[mi][0], b0, acc[mi]);
            acc[mi] = wmma_bf16(a[mi][1], b1, acc[mi]);
        }
#pragma unroll
        for (int mi = 0; mi < 4; ++mi)
#pragma unroll
            for (int r = 0; r < 8; ++r)
                out[(row0 + mi * 16 + rbase + r) * INDIM + col0 + n * 16 + col]
                    = acc[mi][r] + bias;
    }
}

// ---------------------------------------------------------------------------
extern "C" void kernel_launch(void* const* d_in, const int* in_sizes, int n_in,
                              void* d_out, int out_size, void* d_ws, size_t ws_size,
                              hipStream_t stream) {
    const float* x    = (const float*)d_in[0];
    const float* wqkv = (const float*)d_in[1];
    const float* wout = (const float*)d_in[2];
    const float* bout = (const float*)d_in[3];
    float* out = (float*)d_out;

    char* ws = (char*)d_ws;
    size_t off = 0;
    auto alloc = [&](size_t bytes) -> void* {
        void* p = ws + off;
        off = (off + bytes + 255) & ~(size_t)255;
        return p;
    };
    __bf16* wqt  = (__bf16*)alloc((size_t)192 * INDIM * 2);          // 288 KB
    __bf16* wot  = (__bf16*)alloc((size_t)INDIM * DHEAD * 2);        //  96 KB
    __bf16* qws  = (__bf16*)alloc((size_t)NBATCH * NTOK * DHEAD * 2); // 8 MB
    __bf16* kws  = (__bf16*)alloc((size_t)NBATCH * NTOK * DHEAD * 2); // 8 MB
    __bf16* vtws = (__bf16*)alloc((size_t)NBATCH * DHEAD * NTOK * 2); // 8 MB
    __bf16* aws  = (__bf16*)alloc((size_t)NBATCH * NTOK * DHEAD * 2); // 8 MB

    prep_kernel<<<(INDIM * 192 + 255) / 256, 256, 0, stream>>>(wqkv, wout, wqt, wot);
    qkv_kernel<<<256, 256, 0, stream>>>(x, wqt, qws, kws, vtws);     // 2048 waves, M=32
    attn_kernel<<<512, 256, 0, stream>>>(qws, kws, vtws, aws);       // 4096 waves
    out_kernel<<<1536, 256, 0, stream>>>(aws, wot, bout, out);       // 12288 waves, M=64
}
